// Loss_36163624632695
// MI455X (gfx1250) — matrix operand order repaired and checked by reading, not compiled, not run on hardware
//
#include <hip/hip_runtime.h>
#include <math.h>

#define EPS 1e-6f
constexpr int NCLS   = 2048;        // classes / anchors
constexpr int NS     = 5;
constexpr int NQ     = 8;
constexpr int DIM    = 256;
constexpr int SEQ    = NS + NQ;     // 13
constexpr int NQROWS = NCLS * NQ;   // 16384 query rows

// proto_gemm tiling
constexpr int NBLK_N   = 4;                 // N-split across blockIdx.y
constexpr int NCOLS_BLK = NCLS / NBLK_N;    // 512 columns per block
constexpr int NT_BLK   = NCOLS_BLK / 16;    // 32 column tiles per block
constexpr int GTILES   = 2;                 // tiles staged per pipeline stage
constexpr int NSTAGES  = NT_BLK / GTILES;   // 16
// LDS layout for staged B (anchors): padded rows to kill bank conflicts
constexpr int ROW_BYTES  = 528;                      // 512B data + 16B pad (132 dwords)
constexpr int ROW_ELEMS  = ROW_BYTES / 2;            // 264 bf16
constexpr int TILE_BYTES = 16 * ROW_BYTES;           // 8448
constexpr int BUF_BYTES  = GTILES * TILE_BYTES;      // 16896
constexpr int LDS_BYTES  = 2 * BUF_BYTES;            // 33792 (double buffered)

typedef __attribute__((ext_vector_type(16))) __bf16 v16bf;
typedef __attribute__((ext_vector_type(8)))  float  v8f;

union FragAB { v16bf v; uint4 u[2]; };
union FragC  { v8f   v; float f[8]; };

__device__ __forceinline__ void async_b128_to_lds(unsigned lds_off, unsigned voff,
                                                  const void* sbase) {
    // GVS form: LDS dest offset (VGPR), 32-bit byte offset (VGPR), 64-bit base (SGPR pair)
    asm volatile("global_load_async_to_lds_b128 %0, %1, %2 offset:0"
                 :: "v"(lds_off), "v"(voff), "s"(sbase) : "memory");
}
__device__ __forceinline__ void wait_async_le(int n) {
    if (n == 0) asm volatile("s_wait_asynccnt 0x0" ::: "memory");
    else        asm volatile("s_wait_asynccnt 0x4" ::: "memory");
}

// ---------------------------------------------------------------------------
// Prep 1: anchors = mean of 5 supports -> bf16 [2048][256];
//         at[n] = a_sq - 2*EPS*a_sum
// ---------------------------------------------------------------------------
__global__ void __launch_bounds__(256)
prep_anchors(const float* __restrict__ x, __bf16* __restrict__ Abf,
             float* __restrict__ at) {
    const int n = blockIdx.x, d = threadIdx.x;
    const float* base = x + ((size_t)n * SEQ) * DIM + d;
    float a = 0.f;
#pragma unroll
    for (int s = 0; s < NS; ++s) a += base[s * DIM];
    a *= (1.f / NS);
    Abf[(size_t)n * DIM + d] = (__bf16)a;

    __shared__ float s_sq[256], s_sum[256];
    s_sq[d] = a * a; s_sum[d] = a;
    __syncthreads();
    for (int off = 128; off; off >>= 1) {
        if (d < off) { s_sq[d] += s_sq[d + off]; s_sum[d] += s_sum[d + off]; }
        __syncthreads();
    }
    if (d == 0) at[n] = s_sq[0] - 2.f * EPS * s_sum[0];
}

// ---------------------------------------------------------------------------
// Prep 2: queries -> bf16 [16384][256]; qt[q] = q_sq + 2*EPS*q_sum + D*EPS^2
// ---------------------------------------------------------------------------
__global__ void __launch_bounds__(256)
prep_queries(const float* __restrict__ x, __bf16* __restrict__ Qbf,
             float* __restrict__ qt) {
    const int q = blockIdx.x, d = threadIdx.x;
    const int n = q >> 3, j = q & 7;
    const float v = x[((size_t)n * SEQ + NS + j) * DIM + d];
    Qbf[(size_t)q * DIM + d] = (__bf16)v;

    __shared__ float s_sq[256], s_sum[256];
    s_sq[d] = v * v; s_sum[d] = v;
    __syncthreads();
    for (int off = 128; off; off >>= 1) {
        if (d < off) { s_sq[d] += s_sq[d + off]; s_sum[d] += s_sum[d + off]; }
        __syncthreads();
    }
    if (d == 0) qt[q] = s_sq[0] + 2.f * EPS * s_sum[0] + DIM * EPS * EPS;
}

// ---------------------------------------------------------------------------
// Fused WMMA GEMM + online softmax. Block: 8 waves, M=128 (16 rows/wave,
// A register-resident), N=512 columns. B (anchors) streams through LDS via
// double-buffered global_load_async_to_lds_b128 stages (2 tiles = 32 anchors
// per stage), shared by all 8 waves. Per tile, all 8 B fragments are loaded
// into distinct registers first (one DS clause), then 8 WMMAs retire with
// graduated dscnt waits -> DS fetch overlaps the matrix pipe.
// ---------------------------------------------------------------------------
__global__ void __launch_bounds__(256)
proto_gemm(const __bf16* __restrict__ Qbf, const __bf16* __restrict__ Abf,
           const float* __restrict__ qt, const float* __restrict__ at,
           const int* __restrict__ label, float* __restrict__ part) {
    __shared__ __bf16 ldsB[LDS_BYTES / 2];

    const int tid  = threadIdx.x;
    const int wave = tid >> 5;
    const int lane = tid & 31;
    const int hi   = lane >> 4;           // lane half
    const int l16  = lane & 15;
    const int nblk = blockIdx.y;          // N-split index (0..3)
    const int col0 = nblk * NCOLS_BLK;    // first anchor column of this block
    const int mrow0 = (blockIdx.x * 8 + wave) * 16;  // this wave's first query row

    const unsigned lds_base = (unsigned)(uintptr_t)(&ldsB[0]); // LDS byte offset

    // --- A fragments: this wave's 16 rows x 256 K, register resident ------
    FragAB fa[8];
    {
        const __bf16* row = Qbf + (size_t)(mrow0 + l16) * DIM + hi * 8;
#pragma unroll
        for (int kt = 0; kt < 8; ++kt) {
            fa[kt].u[0] = *(const uint4*)(row + kt * 32);
            fa[kt].u[1] = *(const uint4*)(row + kt * 32 + 16);
        }
    }

    // --- per-lane state for 8 C rows (row = mrow0 + hi*8 + r) -------------
    float qloc[8], mx[8], sm[8], bv[8], lv[8];
    int   lcol[8], bi[8];
#pragma unroll
    for (int r = 0; r < 8; ++r) {
        const int m = mrow0 + hi * 8 + r;
        qloc[r] = qt[m];
        lcol[r] = label[m >> 3];
        mx[r] = -INFINITY; sm[r] = 0.f;
        bv[r] = -INFINITY; bi[r] = 0; lv[r] = -INFINITY;
    }

    // --- async-copy issue: 1024 b128 chunks per stage, 4 per thread -------
    auto issue_stage = [&](int s, int buf) {
        const int arow0 = col0 + s * (GTILES * 16); // first anchor row of stage
#pragma unroll
        for (int i = 0; i < 4; ++i) {
            const int ch   = tid + i * 256;   // 0..1023
            const int tile = ch >> 9;         // 0..1
            const int row  = (ch >> 5) & 15;  // 0..15
            const int prt  = ch & 31;         // 0..31 (16B chunks of 512B row)
            const unsigned voff = (unsigned)((arow0 + tile * 16 + row) * 512 + prt * 16);
            const unsigned ldso = lds_base +
                (unsigned)(buf * BUF_BYTES + tile * TILE_BYTES + row * ROW_BYTES + prt * 16);
            async_b128_to_lds(ldso, voff, (const void*)Abf);
        }
    };

    issue_stage(0, 0);
    for (int s = 0; s < NSTAGES; ++s) {
        const int buf = s & 1;
        if (s + 1 < NSTAGES) { issue_stage(s + 1, buf ^ 1); wait_async_le(4); }
        else                 { wait_async_le(0); }
        __syncthreads();                       // stage s staged & visible

#pragma unroll
        for (int t = 0; t < GTILES; ++t) {
            const int nbase = col0 + (s * GTILES + t) * 16;
            const int ncol  = nbase + l16;
            const float a_t = at[ncol];        // hoisted above the WMMA chain

            const __bf16* bp = ldsB + (buf * BUF_BYTES + t * TILE_BYTES) / 2
                               + l16 * ROW_ELEMS + hi * 8;

            // load ALL 8 B fragments into distinct registers (one DS clause)
            FragAB fb[8];
#pragma unroll
            for (int kt = 0; kt < 8; ++kt) {
                fb[kt].u[0] = *(const uint4*)(bp + kt * 32);
                fb[kt].u[1] = *(const uint4*)(bp + kt * 32 + 16);
            }

            FragC c;
#pragma unroll
            for (int r = 0; r < 8; ++r) c.f[r] = 0.f;
#pragma unroll
            for (int kt = 0; kt < 8; ++kt) {
                c.v = __builtin_amdgcn_wmma_f32_16x16x32_bf16(
                    false, fa[kt].v, false, fb[kt].v, (short)0, c.v, false, false);
            }

#pragma unroll
            for (int r = 0; r < 8; ++r) {
                const float l  = 2.f * c.f[r] - qloc[r] - a_t;
                const float nm = fmaxf(mx[r], l);
                sm[r] = sm[r] * __expf(mx[r] - nm) + __expf(l - nm);
                mx[r] = nm;
                if (l > bv[r]) { bv[r] = l; bi[r] = ncol; }
                if (ncol == lcol[r]) lv[r] = l;
            }
        }
        __syncthreads();                       // done reading buf before reuse
    }

    // --- within-wave reduction across the 16 lane-columns of each half ----
#pragma unroll
    for (int off = 8; off; off >>= 1) {
#pragma unroll
        for (int r = 0; r < 8; ++r) {
            const float om = __shfl_xor(mx[r], off, 32);
            const float os = __shfl_xor(sm[r], off, 32);
            const float nm = fmaxf(mx[r], om);
            sm[r] = sm[r] * __expf(mx[r] - nm) + os * __expf(om - nm);
            mx[r] = nm;
            const float ob  = __shfl_xor(bv[r], off, 32);
            const int   obi = __shfl_xor(bi[r], off, 32);
            if (ob > bv[r]) { bv[r] = ob; bi[r] = obi; }
            lv[r] = fmaxf(lv[r], __shfl_xor(lv[r], off, 32));
        }
    }
    if (l16 == 0) {        // lanes 0 and 16 hold reduced stats for 8 rows each
#pragma unroll
        for (int r = 0; r < 8; ++r) {
            const int row = mrow0 + hi * 8 + r;
            float* p = part + ((size_t)row * NBLK_N + nblk) * 8;
            p[0] = mx[r]; p[1] = sm[r]; p[2] = bv[r];
            p[3] = __int_as_float(bi[r]); p[4] = lv[r];
        }
    }
}

// ---------------------------------------------------------------------------
// Merge the NBLK_N column-split partials per row -> loss / accuracy sums.
// ---------------------------------------------------------------------------
__global__ void __launch_bounds__(256)
merge_rows(const float* __restrict__ part, const int* __restrict__ label,
           float* __restrict__ acc) {
    const int row = blockIdx.x * 256 + threadIdx.x;
    float gm = -INFINITY, gs = 0.f, gb = -INFINITY, gl = -INFINITY;
    int   gbi = 0;
#pragma unroll
    for (int p = 0; p < NBLK_N; ++p) {
        const float* q = part + ((size_t)row * NBLK_N + p) * 8;
        const float m_i = q[0];
        const float nm  = fmaxf(gm, m_i);
        gs = gs * __expf(gm - nm) + q[1] * __expf(m_i - nm);
        gm = nm;
        if (q[2] > gb) { gb = q[2]; gbi = __float_as_int(q[3]); }
        gl = fmaxf(gl, q[4]);
    }
    const float loss = -(gl - (gm + __logf(gs)));
    const float corr = (gbi == label[row >> 3]) ? 1.f : 0.f;

    __shared__ float s_l[256], s_c[256];
    s_l[threadIdx.x] = loss; s_c[threadIdx.x] = corr;
    __syncthreads();
    for (int off = 128; off; off >>= 1) {
        if (threadIdx.x < off) {
            s_l[threadIdx.x] += s_l[threadIdx.x + off];
            s_c[threadIdx.x] += s_c[threadIdx.x + off];
        }
        __syncthreads();
    }
    if (threadIdx.x == 0) { atomicAdd(&acc[0], s_l[0]); atomicAdd(&acc[1], s_c[0]); }
}

__global__ void finalize(const float* __restrict__ acc, float* __restrict__ out) {
    out[0] = acc[0] / (float)NQROWS;
    out[1] = acc[1] / (float)NQROWS * 100.f;
}

// ---------------------------------------------------------------------------
extern "C" void kernel_launch(void* const* d_in, const int* in_sizes, int n_in,
                              void* d_out, int out_size, void* d_ws, size_t ws_size,
                              hipStream_t stream) {
    const float* x     = (const float*)d_in[0];
    const int*   label = (const int*)d_in[1];

    char* ws = (char*)d_ws;
    __bf16* Abf  = (__bf16*)ws;                                  // 1 MB
    __bf16* Qbf  = (__bf16*)(ws + (1u << 20));                   // 8 MB
    float*  at   = (float*)(ws + (9u << 20));                    // 8 KB
    float*  qt   = at + NCLS;                                    // 64 KB
    float*  part = qt + NQROWS;                                  // 2 MB partials
    float*  acc  = part + (size_t)NQROWS * NBLK_N * 8;           // 8 B

    hipMemsetAsync(acc, 0, 2 * sizeof(float), stream);
    prep_anchors<<<NCLS,   256, 0, stream>>>(x, Abf, at);
    prep_queries<<<NQROWS, 256, 0, stream>>>(x, Qbf, qt);
    proto_gemm<<<dim3(NQROWS / 128, NBLK_N), 256, 0, stream>>>(Qbf, Abf, qt, at, label, part);
    merge_rows<<<NQROWS / 256, 256, 0, stream>>>(part, label, acc);
    finalize<<<1, 1, 0, stream>>>(acc, (float*)d_out);
}
